// SendRecv_33767032881805
// MI455X (gfx1250) — compile-verified, matrix operand order checked by reading, not run on hardware
//
#include <hip/hip_runtime.h>

#define BT       65536      // B*T
#define IN_DIM   256
#define DLAT     64
#define KCODE    1024
#define ROWS_PB  128        // tokens per block
#define THREADS  256        // 8 wave32
#define NBLOCKS  (BT / ROWS_PB)

typedef __attribute__((ext_vector_type(16))) __bf16 v16bf;
typedef __attribute__((ext_vector_type(2)))  __bf16 v2bf;
typedef __attribute__((ext_vector_type(8)))  float  v8f;
typedef __attribute__((ext_vector_type(4)))  unsigned int u32x4;
typedef __attribute__((ext_vector_type(8)))  unsigned int u32x8;

union Frag { v16bf v; unsigned int u[8]; };

__device__ __forceinline__ unsigned short f2bf(float f) {
  __bf16 h = (__bf16)f;                       // native cvt (RTNE)
  return __builtin_bit_cast(unsigned short, h);
}
__device__ __forceinline__ float bf2f(unsigned short h) {
  return __uint_as_float(((unsigned int)h) << 16);
}
__device__ __forceinline__ unsigned int pk2(float a, float b) {
  v2bf p; p.x = (__bf16)a; p.y = (__bf16)b;   // -> v_cvt_pk_bf16_f32 class
  return __builtin_bit_cast(unsigned int, p);
}
// 16-bit WMMA fragment: per lane, K = kb..kb+7 and kb+16..kb+23 are contiguous
__device__ __forceinline__ void ldfrag(Frag& f, const unsigned short* base) {
  const uint4* p = (const uint4*)base;        // base = &arr[row*stride + kb]
  uint4 lo = p[0];                            // 8 bf16: k .. k+7
  uint4 hi = p[2];                            // 8 bf16: k+16 .. k+23
  f.u[0] = lo.x; f.u[1] = lo.y; f.u[2] = lo.z; f.u[3] = lo.w;
  f.u[4] = hi.x; f.u[5] = hi.y; f.u[6] = hi.z; f.u[7] = hi.w;
}

extern "C" __global__ void __launch_bounds__(THREADS)
vq_init(unsigned int* counts, float* loss_out) {
  int t = blockIdx.x * blockDim.x + threadIdx.x;
  if (t < KCODE) counts[t] = 0u;
  if (t < 2) loss_out[t] = 0.0f;
}

extern "C" __global__ void __launch_bounds__(THREADS)
vq_entropy(const unsigned int* counts, float* ent_out) {
  __shared__ float red[THREADS];
  int t = threadIdx.x;
  float s = 0.0f;
  for (int k = t; k < KCODE; k += THREADS) {
    unsigned int c = counts[k];
    if (c > 0u) {
      float p = (float)c * (1.0f / (float)BT);
      s += p * __logf(p);
    }
  }
  red[t] = s;
  __syncthreads();
  for (int o = THREADS / 2; o > 0; o >>= 1) {
    if (t < o) red[t] += red[t + o];
    __syncthreads();
  }
  if (t == 0) *ent_out = -red[0] * 1.4426950408889634f;  // /ln2
}

extern "C" __global__ void __launch_bounds__(THREADS)
vqvae_fused(const float* __restrict__ x, const float* __restrict__ send_w,
            const float* __restrict__ send_b, const float* __restrict__ emb,
            const float* __restrict__ recv_w, const float* __restrict__ recv_b,
            float* __restrict__ xhat, float* __restrict__ codes_out,
            float* __restrict__ loss_out, unsigned int* __restrict__ counts) {
  extern __shared__ unsigned short smem[];
  unsigned short* sw = smem;                      // send_w^T  [64][256] bf16
  unsigned short* eb = sw + DLAT * IN_DIM;        // emb       [1024][64] bf16
  unsigned short* rw = eb + KCODE * DLAT;         // recv_w^T  [256][64] bf16
  unsigned short* zt = rw + IN_DIM * DLAT;        // z tile    [128][64] bf16
  unsigned short* vt = zt + ROWS_PB * DLAT;       // vecs tile [128][64] bf16
  float* sqr   = (float*)(vt + ROWS_PB * DLAT);   // ||e_k||^2 [1024] f32
  float* swf32 = sqr + KCODE;                     // TDM stage: send_w f32 [256*64]

  const int tid = threadIdx.x;

  // ---- TDM: DMA send_w (64KB, contiguous f32) global -> LDS ----
  if (tid < 32) {   // one wave issues; TDM ignores EXEC
    unsigned lds_off = (unsigned)(size_t)(void*)swf32;       // LDS aperture: low 32b = offset
    unsigned long long ga = (unsigned long long)(size_t)send_w;
    u32x4 g0;
    g0.x = 1u;                                               // count=1, valid descriptor
    g0.y = lds_off;                                          // lds_addr
    g0.z = (unsigned)ga;                                     // global_addr[31:0]
    g0.w = (unsigned)((ga >> 32) & 0x1FFFFFFu) | 0x80000000u; // addr[56:32] | type=2
    u32x8 g1;
    g1.s0 = 0x20000u;      // data_size=2 (4B), no mask/pad/iterate
    g1.s1 = 0x40000000u;   // tensor_dim0[15:0]=16384 in [31:16]
    g1.s2 = 0x00010000u;   // tensor_dim1=1
    g1.s3 = 0x40000000u;   // tile_dim0=16384 in [31:16]
    g1.s4 = 1u;            // tile_dim1=1, tile_dim2=0
    g1.s5 = 16384u;        // tensor_dim0_stride
    g1.s6 = 0u;
    g1.s7 = 0u;
    asm volatile("tensor_load_to_lds %0, %1" :: "s"(g0), "s"(g1) : "memory");
    __builtin_amdgcn_s_wait_tensorcnt(0);
  }
  __syncthreads();

  // ---------------- stage weights into LDS (f32 -> bf16) ----------------
  for (int e = tid; e < IN_DIM * DLAT; e += THREADS) {   // swf32 [k][n] -> sw[n][k]
    int k = e / DLAT, n = e % DLAT;
    sw[n * IN_DIM + k] = f2bf(swf32[e]);
  }
  for (int e = tid; e < DLAT * IN_DIM; e += THREADS) {   // recv_w [d][n] -> rw[n][d]
    int d = e / IN_DIM, n = e % IN_DIM;
    rw[n * DLAT + d] = f2bf(recv_w[e]);
  }
  for (int row = tid; row < KCODE; row += THREADS) {     // emb row-major + row norms
    const float4* er = (const float4*)(emb + row * DLAT);
    unsigned int* ew = (unsigned int*)(eb + row * DLAT);
    float s = 0.0f;
#pragma unroll
    for (int i = 0; i < DLAT / 4; ++i) {
      float4 v = er[i];
      s += v.x * v.x + v.y * v.y + v.z * v.z + v.w * v.w;
      ew[2 * i]     = pk2(v.x, v.y);
      ew[2 * i + 1] = pk2(v.z, v.w);
    }
    sqr[row] = s;
  }
  __syncthreads();

  const int wave = tid >> 5;
  const int lane = tid & 31;
  const int half = lane >> 4;      // K-chunk half per WMMA 16-bit layout
  const int l16  = lane & 15;
  const int r0   = wave * 16;      // this wave's local row base
  const long rowG0 = (long)blockIdx.x * ROWS_PB + r0;

  // ---------------- GEMM1: z[16x64] = x[16x256] @ send_w ----------------
  v8f zacc[4];
#pragma unroll
  for (int nt = 0; nt < 4; ++nt) zacc[nt] = (v8f){};
  const float* xrow = x + (rowG0 + l16) * IN_DIM;
  __builtin_prefetch(xrow, 0, 3);                     // global_prefetch_b8
#pragma unroll 1
  for (int kk = 0; kk < IN_DIM / 32; ++kk) {
    int kb = kk * 32 + half * 8;
    Frag a;
    {
      const float4* xp = (const float4*)(xrow + kb);
      float4 f0 = xp[0], f1 = xp[1], f2 = xp[4], f3 = xp[5];  // k..k+7, k+16..k+23
      a.u[0] = pk2(f0.x, f0.y); a.u[1] = pk2(f0.z, f0.w);
      a.u[2] = pk2(f1.x, f1.y); a.u[3] = pk2(f1.z, f1.w);
      a.u[4] = pk2(f2.x, f2.y); a.u[5] = pk2(f2.z, f2.w);
      a.u[6] = pk2(f3.x, f3.y); a.u[7] = pk2(f3.z, f3.w);
    }
#pragma unroll
    for (int nt = 0; nt < 4; ++nt) {
      Frag b;
      ldfrag(b, sw + (nt * 16 + l16) * IN_DIM + kb);
      zacc[nt] = __builtin_amdgcn_wmma_f32_16x16x32_bf16(
          false, a.v, false, b.v, (short)0, zacc[nt], false, false);
    }
  }
  // bias + spill z to LDS (bf16) for GEMM2 A fragments
#pragma unroll
  for (int nt = 0; nt < 4; ++nt) {
    float bn = send_b[nt * 16 + l16];
#pragma unroll
    for (int r = 0; r < 8; ++r) {
      zacc[nt][r] += bn;
      int m = r + half * 8;
      zt[(r0 + m) * DLAT + nt * 16 + l16] = f2bf(zacc[nt][r]);
    }
  }

  // ---------------- GEMM2 + running argmin over K=1024 codewords --------
  Frag za0, za1;
  ldfrag(za0, zt + (r0 + l16) * DLAT + half * 8);
  ldfrag(za1, zt + (r0 + l16) * DLAT + 32 + half * 8);
  float bestV[8]; int bestI[8];
#pragma unroll
  for (int r = 0; r < 8; ++r) { bestV[r] = 3.4e38f; bestI[r] = 0x7fffffff; }
#pragma unroll 1
  for (int nt = 0; nt < KCODE / 16; ++nt) {
    int n = nt * 16 + l16;
    Frag b0, b1;
    ldfrag(b0, eb + n * DLAT + half * 8);
    ldfrag(b1, eb + n * DLAT + 32 + half * 8);
    v8f c = {};
    c = __builtin_amdgcn_wmma_f32_16x16x32_bf16(false, za0.v, false, b0.v, (short)0, c, false, false);
    c = __builtin_amdgcn_wmma_f32_16x16x32_bf16(false, za1.v, false, b1.v, (short)0, c, false, false);
    float sq = sqr[n];
#pragma unroll
    for (int r = 0; r < 8; ++r) {
      float score = sq - 2.0f * c[r];
      if (score < bestV[r] || (score == bestV[r] && n < bestI[r])) {
        bestV[r] = score; bestI[r] = n;
      }
    }
  }
  // finish argmin across the 16 lanes of each half (rows m and m+8)
#pragma unroll
  for (int s = 1; s < 16; s <<= 1) {
#pragma unroll
    for (int r = 0; r < 8; ++r) {
      float ov = __shfl_xor(bestV[r], s, 32);
      int   oi = __shfl_xor(bestI[r], s, 32);
      if (ov < bestV[r] || (ov == bestV[r] && oi < bestI[r])) {
        bestV[r] = ov; bestI[r] = oi;
      }
    }
  }
  if (l16 == 0) {
#pragma unroll
    for (int r = 0; r < 8; ++r) {
      int m = r + half * 8;
      codes_out[rowG0 + m] = (float)bestI[r];
      atomicAdd(&counts[bestI[r]], 1u);
    }
  }
  // gather vecs = emb[codes] into vt (LDS -> LDS, 2 dwords/lane/row)
#pragma unroll
  for (int r = 0; r < 8; ++r) {
    int m = r + half * 8;
    const unsigned int* src = (const unsigned int*)(eb + bestI[r] * DLAT);
    unsigned int* dst = (unsigned int*)(vt + (r0 + m) * DLAT);
    dst[l16]      = src[l16];
    dst[l16 + 16] = src[l16 + 16];
  }

  // ---------------- losses: sum((z - vecs)^2), same forward value twice --
  float loss = 0.0f;
#pragma unroll
  for (int nt = 0; nt < 4; ++nt) {
#pragma unroll
    for (int r = 0; r < 8; ++r) {
      int m = r + half * 8;
      float dv = zacc[nt][r] - bf2f(vt[(r0 + m) * DLAT + nt * 16 + l16]);
      loss += dv * dv;
    }
  }
#pragma unroll
  for (int s = 1; s < 32; s <<= 1) loss += __shfl_xor(loss, s, 32);
  if (lane == 0) {
    unsafeAtomicAdd(&loss_out[0], loss);   // embedding
    unsafeAtomicAdd(&loss_out[1], loss);   // commitment (same forward value)
  }

  // ---------------- GEMM3: x_hat[16x256] = vecs[16x64] @ recv_w + b -----
  Frag va0, va1;
  ldfrag(va0, vt + (r0 + l16) * DLAT + half * 8);
  ldfrag(va1, vt + (r0 + l16) * DLAT + 32 + half * 8);
#pragma unroll 1
  for (int nt = 0; nt < IN_DIM / 16; ++nt) {
    int n = nt * 16 + l16;
    Frag b0, b1;
    ldfrag(b0, rw + n * DLAT + half * 8);
    ldfrag(b1, rw + n * DLAT + 32 + half * 8);
    v8f c = {};
    c = __builtin_amdgcn_wmma_f32_16x16x32_bf16(false, va0.v, false, b0.v, (short)0, c, false, false);
    c = __builtin_amdgcn_wmma_f32_16x16x32_bf16(false, va1.v, false, b1.v, (short)0, c, false, false);
    float rb = recv_b[n];
#pragma unroll
    for (int r = 0; r < 8; ++r) {
      int m = r + half * 8;
      xhat[(rowG0 + m) * IN_DIM + n] = c[r] + rb;
    }
  }
}

extern "C" void kernel_launch(void* const* d_in, const int* in_sizes, int n_in,
                              void* d_out, int out_size, void* d_ws, size_t ws_size,
                              hipStream_t stream) {
  const float* x      = (const float*)d_in[0];
  const float* send_w = (const float*)d_in[1];
  const float* send_b = (const float*)d_in[2];
  const float* emb    = (const float*)d_in[3];
  const float* recv_w = (const float*)d_in[4];
  const float* recv_b = (const float*)d_in[5];

  float* out    = (float*)d_out;
  float* xhat   = out;                              // [BT, IN_DIM]
  float* codes  = out + (size_t)BT * IN_DIM;        // [BT]
  float* losses = codes + BT;                       // [embedding, commitment, ent]
  unsigned int* counts = (unsigned int*)d_ws;       // [KCODE]

  size_t smem_bytes =
      (size_t)(DLAT * IN_DIM + KCODE * DLAT + IN_DIM * DLAT +
               ROWS_PB * DLAT + ROWS_PB * DLAT) * sizeof(unsigned short) +
      (size_t)KCODE * sizeof(float) +
      (size_t)IN_DIM * DLAT * sizeof(float);        // ~292 KB < 320 KB WGP LDS

  hipFuncSetAttribute(reinterpret_cast<const void*>(vqvae_fused),
                      hipFuncAttributeMaxDynamicSharedMemorySize, (int)smem_bytes);

  vq_init<<<(KCODE + THREADS - 1) / THREADS, THREADS, 0, stream>>>(counts, losses);
  vqvae_fused<<<NBLOCKS, THREADS, smem_bytes, stream>>>(
      x, send_w, send_b, emb, recv_w, recv_b, xhat, codes, losses, counts);
  vq_entropy<<<1, THREADS, 0, stream>>>(counts, losses + 2);
}
